// SimpleMoEModel_64725157151313
// MI455X (gfx1250) — matrix operand order repaired and checked by reading, not compiled, not run on hardware
//
#include <hip/hip_runtime.h>
#include <hip/hip_bf16.h>

// MoE forward on gfx1250 (MI455X).
// Heavy GEMMs: v_wmma_f32_16x16x32_bf16, operands pre-converted to bf16 once,
// tiles staged global->LDS with gfx1250 global_load_async_to_lds_b128 (ASYNCcnt).
// Workspace (bytes): h32 64M | eo 64M | o32 64M | xb 32M | dispb 32M |
//                    W1t 2M | We2t 16M | We3t 16M | idx/slot/gate/sent ~0.2M
// => requires ws_size >= ~291 MB.

#define TOKENS 16384
#define DMODEL 1024
#define NEXP   8
#define CAP    2048

typedef __attribute__((ext_vector_type(16))) __bf16 v16bf;
typedef __attribute__((ext_vector_type(8)))  __bf16 bf16x8;
typedef __attribute__((ext_vector_type(4)))  __bf16 bf16x4;
typedef __attribute__((ext_vector_type(2)))  __bf16 bf16x2;
typedef __attribute__((ext_vector_type(8)))  float  v8f;

static __device__ __forceinline__ __bf16 f2bf(float f) {
  union { float f; unsigned u; } v; v.f = f;
  unsigned r = v.u + 0x7FFFu + ((v.u >> 16) & 1u);   // round-to-nearest-even
  unsigned short h = (unsigned short)(r >> 16);
  __bf16 o; __builtin_memcpy(&o, &h, sizeof(o));
  return o;
}

static __device__ __forceinline__ bf16x2 pk2bf(float a, float b) {
#if defined(__has_builtin)
#if __has_builtin(__builtin_amdgcn_cvt_pk_bf16_f32)
  auto r = __builtin_amdgcn_cvt_pk_bf16_f32(a, b);   // packed hw convert
  bf16x2 o; __builtin_memcpy(&o, &r, sizeof(o));
  return o;
#else
  bf16x2 o; o[0] = f2bf(a); o[1] = f2bf(b); return o;
#endif
#else
  bf16x2 o; o[0] = f2bf(a); o[1] = f2bf(b); return o;
#endif
}

// ---------------------------------------------------------------------------
// Tiled WMMA GEMM: C[z](Mx1024) = A[z](Mx1024,bf16) @ B[z] + bias[z].
// B supplied PRE-TRANSPOSED bf16: Bt[n][k]. Block tile 128x128, 256 thr =
// 8 waves (2Mx4N), wave tile 64x32 = 4x2 16x16 frags. K-step 32, double-
// buffered LDS filled with global_load_async_to_lds_b128 (no staging VALU).
// ---------------------------------------------------------------------------
__global__ __launch_bounds__(256)
void gemm_bf16_kernel(const __bf16* __restrict__ Ab, const __bf16* __restrict__ Btb,
                      const float* __restrict__ biasb, float* __restrict__ Cb,
                      long long sA, long long sB, long long sBias, long long sC)
{
  const int z = blockIdx.z;
  const __bf16* A    = Ab  + (long long)z * sA;
  const __bf16* Bt   = Btb + (long long)z * sB;
  const float*  bias = biasb + (long long)z * sBias;
  float*        C    = Cb  + (long long)z * sC;

  const int tid  = threadIdx.x;
  const int lane = tid & 31;
  const int wave = tid >> 5;
  const int wm   = wave & 1;          // 2 wave rows    -> 64 M each
  const int wn   = wave >> 1;         // 4 wave columns -> 32 N each
  const int l15  = lane & 15;
  const int lh   = lane >> 4;
  const int mBase = blockIdx.y * 128;
  const int nBase = blockIdx.x * 128;

  // A tile [128 rows][K=32 (+8 pad)], B tile [128 cols][K=32 (+8 pad)]
  __shared__ __align__(16) __bf16 sAt[2][128 * 40];
  __shared__ __align__(16) __bf16 sBt[2][128 * 40];

  // Staging map: tile = 128 rows x 64B = 512 16B-chunks; 2 chunks/thread.
  const int r0 = tid >> 2,        q0 = (tid & 3) << 3;          // chunk c=tid
  const int r1 = (tid + 256) >> 2, q1 = ((tid + 256) & 3) << 3; // chunk c=tid+256

  v8f zero8 = {0.f,0.f,0.f,0.f,0.f,0.f,0.f,0.f};
  v8f acc[4][2];
#pragma unroll
  for (int i = 0; i < 4; ++i)
#pragma unroll
    for (int j = 0; j < 2; ++j) acc[i][j] = zero8;

  // async 16B copy: global (bf16, K-contiguous) -> LDS
#define ASYNC16(ldsp, gp)                                                     \
  { unsigned _l = (unsigned)(size_t)(ldsp);                                   \
    asm volatile("global_load_async_to_lds_b128 %0, %1, off"                  \
                 :: "v"(_l), "v"(gp) : "memory"); }

  // ---- prologue: stage K-tile 0 into buffer 0 ----
  {
    ASYNC16(&sAt[0][r0 * 40 + q0], A  + (long long)(mBase + r0) * 1024 + q0);
    ASYNC16(&sAt[0][r1 * 40 + q1], A  + (long long)(mBase + r1) * 1024 + q1);
    ASYNC16(&sBt[0][r0 * 40 + q0], Bt + (long long)(nBase + r0) * 1024 + q0);
    ASYNC16(&sBt[0][r1 * 40 + q1], Bt + (long long)(nBase + r1) * 1024 + q1);
  }
  asm volatile("s_wait_asynccnt 0x0" ::: "memory");
  __syncthreads();

  for (int kt = 0; kt < 32; ++kt) {
    const int cur = kt & 1;
    if (kt + 1 < 32) {                       // async-prefetch next K tile
      const int nb = cur ^ 1;
      const int kk = (kt + 1) * 32;
      ASYNC16(&sAt[nb][r0 * 40 + q0], A  + (long long)(mBase + r0) * 1024 + kk + q0);
      ASYNC16(&sAt[nb][r1 * 40 + q1], A  + (long long)(mBase + r1) * 1024 + kk + q1);
      ASYNC16(&sBt[nb][r0 * 40 + q0], Bt + (long long)(nBase + r0) * 1024 + kk + q0);
      ASYNC16(&sBt[nb][r1 * 40 + q1], Bt + (long long)(nBase + r1) * 1024 + kk + q1);
    }

    // ---- fragment loads from LDS (layouts per CDNA5 ISA 7.12.2) ----
    const __bf16* sa = sAt[cur];
    const __bf16* sb = sBt[cur];
    v16bf af[4], bfr[2];
#pragma unroll
    for (int i = 0; i < 4; ++i) {
      const __bf16* p = sa + (wm * 64 + 16 * i + l15) * 40;
      bf16x8 lo = *(const bf16x8*)(p + 8 * lh);        // K = 8h .. 8h+7
      bf16x8 hi = *(const bf16x8*)(p + 16 + 8 * lh);   // K = 16+8h ..
      af[i] = __builtin_shufflevector(lo, hi, 0,1,2,3,4,5,6,7,8,9,10,11,12,13,14,15);
    }
#pragma unroll
    for (int j = 0; j < 2; ++j) {
      const __bf16* p = sb + (wn * 32 + 16 * j + l15) * 40; // K = 8h..8h+7, 16+8h..
      bf16x8 lo = *(const bf16x8*)(p + 8 * lh);
      bf16x8 hi = *(const bf16x8*)(p + 16 + 8 * lh);
      bfr[j] = __builtin_shufflevector(lo, hi, 0,1,2,3,4,5,6,7,8,9,10,11,12,13,14,15);
    }
#pragma unroll
    for (int i = 0; i < 4; ++i)
#pragma unroll
      for (int j = 0; j < 2; ++j)
        acc[i][j] = __builtin_amdgcn_wmma_f32_16x16x32_bf16(
            false, af[i], false, bfr[j], (short)0, acc[i][j], false, false);

    asm volatile("s_wait_asynccnt 0x0" ::: "memory");
    __syncthreads();
  }
#undef ASYNC16

  // ---- epilogue: bias add + store (C layout: VGPR r -> M = r + 8h, N = l15) ----
#pragma unroll
  for (int j = 0; j < 2; ++j) {
    int n = nBase + wn * 32 + 16 * j + l15;
    float bv = bias[n];
#pragma unroll
    for (int i = 0; i < 4; ++i) {
      int mrow = mBase + wm * 64 + 16 * i + 8 * lh;
      float* cp = C + (long long)mrow * 1024 + n;
#pragma unroll
      for (int r = 0; r < 8; ++r)
        cp[(long long)r * 1024] = acc[i][j][r] + bv;
    }
  }
}

// ---------------------------------------------------------------------------
// Pre-pass: f32 -> bf16 streaming convert (8 elements / thread).
// ---------------------------------------------------------------------------
__global__ __launch_bounds__(256)
void f32_to_bf16_kernel(const float* __restrict__ in, __bf16* __restrict__ out)
{
  size_t i = ((size_t)blockIdx.x * 256 + threadIdx.x) * 8;
  float4 a = *(const float4*)(in + i);
  float4 b = *(const float4*)(in + i + 4);
  bf16x2 p0 = pk2bf(a.x, a.y), p1 = pk2bf(a.z, a.w);
  bf16x2 p2 = pk2bf(b.x, b.y), p3 = pk2bf(b.z, b.w);
  bf16x8 o = { p0[0],p0[1],p1[0],p1[1],p2[0],p2[1],p3[0],p3[1] };
  *(bf16x8*)(out + i) = o;
}

// Pre-pass: W[z] (1024x1024 f32, [k][n]) -> Wt[z] (bf16, [n][k]); 32x32 tiles.
__global__ __launch_bounds__(256)
void transpose_w_kernel(const float* __restrict__ W, __bf16* __restrict__ Wt)
{
  __shared__ float t[32][33];
  const float* Wz  = W  + (size_t)blockIdx.z * DMODEL * DMODEL;
  __bf16*      Wtz = Wt + (size_t)blockIdx.z * DMODEL * DMODEL;
  const int kBase = blockIdx.y * 32, nBase = blockIdx.x * 32;
  const int r = threadIdx.x >> 3, c4 = (threadIdx.x & 7) << 2;
  float4 v = *(const float4*)(Wz + (size_t)(kBase + r) * DMODEL + nBase + c4);
  t[r][c4 + 0] = v.x; t[r][c4 + 1] = v.y; t[r][c4 + 2] = v.z; t[r][c4 + 3] = v.w;
  __syncthreads();
  bf16x2 p0 = pk2bf(t[c4 + 0][r], t[c4 + 1][r]);
  bf16x2 p1 = pk2bf(t[c4 + 2][r], t[c4 + 3][r]);
  bf16x4 o = { p0[0], p0[1], p1[0], p1[1] };
  *(bf16x4*)(Wtz + (size_t)(nBase + r) * DMODEL + kBase + c4) = o;
}

// ---------------------------------------------------------------------------
// Top-1 gating: one wave32 per token; softmax(x @ Wg) -> argmax prob + index.
// ---------------------------------------------------------------------------
__global__ __launch_bounds__(256)
void gate_kernel(const float* __restrict__ X, const float* __restrict__ Wg,
                 int* __restrict__ idx, float* __restrict__ gate)
{
  const int t = blockIdx.x * 8 + (threadIdx.x >> 5);
  const int lane = threadIdx.x & 31;
  const float* row = X + (long long)t * DMODEL;
  float acc[NEXP] = {0.f,0.f,0.f,0.f,0.f,0.f,0.f,0.f};
  for (int d = lane; d < DMODEL; d += 32) {
    float xv = row[d];
    const float* wp = Wg + d * NEXP;
#pragma unroll
    for (int e = 0; e < NEXP; ++e) acc[e] += xv * wp[e];
  }
#pragma unroll
  for (int off = 16; off > 0; off >>= 1)
#pragma unroll
    for (int e = 0; e < NEXP; ++e) acc[e] += __shfl_down(acc[e], off, 32);
  if (lane == 0) {
    float mx = acc[0]; int am = 0;
#pragma unroll
    for (int e = 1; e < NEXP; ++e) if (acc[e] > mx) { mx = acc[e]; am = e; }
    float s = 0.f;
#pragma unroll
    for (int e = 0; e < NEXP; ++e) s += __expf(acc[e] - mx);
    idx[t]  = am;
    gate[t] = 1.f / s;          // softmax probability of the selected expert
  }
}

// Ordered per-expert capacity scan (exact DeepSpeed drop semantics).
__global__ __launch_bounds__(256)
void scan_kernel(const int* __restrict__ idx, int* __restrict__ slot,
                 float* __restrict__ gate)
{
  __shared__ int cnt[256][NEXP];
  const int tid = threadIdx.x;
  const int base = tid * 64;
  int local[NEXP] = {0,0,0,0,0,0,0,0};
  for (int i = 0; i < 64; ++i) local[idx[base + i]]++;
#pragma unroll
  for (int e = 0; e < NEXP; ++e) cnt[tid][e] = local[e];
  __syncthreads();
  if (tid < NEXP) {
    int tot = 0;
    for (int i = 0; i < 256; ++i) { int v = cnt[i][tid]; cnt[i][tid] = tot; tot += v; }
  }
  __syncthreads();
  int off[NEXP];
#pragma unroll
  for (int e = 0; e < NEXP; ++e) off[e] = cnt[tid][e];
  for (int i = 0; i < 64; ++i) {
    int t = base + i, e = idx[t];
    int loc = off[e]++;
    if (loc < CAP) slot[t] = loc;
    else { slot[t] = -1; gate[t] = 0.f; }
  }
}

__global__ __launch_bounds__(256)
void zero_kernel(uint4* __restrict__ p)
{
  p[(size_t)blockIdx.x * 256 + threadIdx.x] = uint4{0u, 0u, 0u, 0u};
}

// Scatter kept tokens into [E, C, D] expert buffer, converting f32 -> bf16.
__global__ __launch_bounds__(128)
void dispatch_kernel(const float* __restrict__ X, const int* __restrict__ idx,
                     const int* __restrict__ slot, __bf16* __restrict__ disp)
{
  const int t = blockIdx.x;
  const int s = slot[t];
  if (s < 0) return;
  const float4* src = (const float4*)(X + (size_t)t * DMODEL);
  bf16x4* dst = (bf16x4*)(disp + ((size_t)idx[t] * CAP + s) * DMODEL);
  for (int i = threadIdx.x; i < DMODEL / 4; i += 128) {
    float4 v = src[i];
    bf16x2 p0 = pk2bf(v.x, v.y), p1 = pk2bf(v.z, v.w);
    dst[i] = bf16x4{p0[0], p0[1], p1[0], p1[1]};
  }
}

__global__ __launch_bounds__(128)
void combine_kernel(const float* __restrict__ eo, const int* __restrict__ idx,
                    const int* __restrict__ slot, const float* __restrict__ gate,
                    float* __restrict__ o)
{
  const int t = blockIdx.x;
  const int s = slot[t];
  float4* dst = (float4*)(o + (size_t)t * DMODEL);
  if (s < 0) {
    float4 zv = {0.f, 0.f, 0.f, 0.f};
    for (int i = threadIdx.x; i < DMODEL / 4; i += 128) dst[i] = zv;
  } else {
    const float g = gate[t];
    const float4* src = (const float4*)(eo + ((size_t)idx[t] * CAP + s) * DMODEL);
    for (int i = threadIdx.x; i < DMODEL / 4; i += 128) {
      float4 v = src[i];
      dst[i] = float4{v.x * g, v.y * g, v.z * g, v.w * g};
    }
  }
}

// sent[b,d] = mean_s(h[b,s,d] + o[b,s,d])
__global__ __launch_bounds__(1024)
void mean_kernel(const float* __restrict__ h, const float* __restrict__ o,
                 float* __restrict__ sent)
{
  const int b = blockIdx.x, d = threadIdx.x;
  const float* hp = h + (size_t)b * 2048 * DMODEL + d;
  const float* op = o + (size_t)b * 2048 * DMODEL + d;
  float acc = 0.f;
  for (int s = 0; s < 2048; ++s) acc += hp[(size_t)s * DMODEL] + op[(size_t)s * DMODEL];
  sent[b * DMODEL + d] = acc * (1.f / 2048.f);
}

// loss = -(1/8) * sum_b (sent[b, y[b]] - logsumexp(sent[b, :]))
__global__ __launch_bounds__(1024)
void loss_kernel(const float* __restrict__ sent, const int* __restrict__ y,
                 float* __restrict__ out)
{
  __shared__ float red[1024];
  __shared__ float tgt;
  const int tid = threadIdx.x;
  float total = 0.f;
  for (int b = 0; b < 8; ++b) {
    float v = sent[b * DMODEL + tid];
    red[tid] = v;
    __syncthreads();
    for (int off = 512; off > 0; off >>= 1) {
      if (tid < off) red[tid] = fmaxf(red[tid], red[tid + off]);
      __syncthreads();
    }
    float mx = red[0];
    __syncthreads();
    red[tid] = __expf(v - mx);
    __syncthreads();
    for (int off = 512; off > 0; off >>= 1) {
      if (tid < off) red[tid] += red[tid + off];
      __syncthreads();
    }
    float lse = mx + __logf(red[0]);
    if (tid == y[b]) tgt = v;
    __syncthreads();
    if (tid == 0) total += tgt - lse;
    __syncthreads();
  }
  if (tid == 0) out[0] = -total * 0.125f;
}

// ---------------------------------------------------------------------------
static void run_moe_layer(const float* xin, const float* Wg, const __bf16* Wet,
                          const float* be, __bf16* dispb, float* eo, float* oout,
                          int* idxb, int* slotb, float* gateb, hipStream_t s)
{
  gate_kernel<<<TOKENS / 8, 256, 0, s>>>(xin, Wg, idxb, gateb);
  scan_kernel<<<1, 256, 0, s>>>(idxb, slotb, gateb);
  // zero 32MB bf16 dispatch buffer (2M uint4)
  zero_kernel<<<(TOKENS * (size_t)DMODEL * 2 / 16) / 256, 256, 0, s>>>((uint4*)dispb);
  dispatch_kernel<<<TOKENS, 128, 0, s>>>(xin, idxb, slotb, dispb);
  gemm_bf16_kernel<<<dim3(8, CAP / 128, NEXP), 256, 0, s>>>(
      dispb, Wet, be, eo,
      (long long)CAP * DMODEL, (long long)DMODEL * DMODEL,
      (long long)DMODEL, (long long)CAP * DMODEL);
  combine_kernel<<<TOKENS, 128, 0, s>>>(eo, idxb, slotb, gateb, oout);
}

extern "C" void kernel_launch(void* const* d_in, const int* in_sizes, int n_in,
                              void* d_out, int out_size, void* d_ws, size_t ws_size,
                              hipStream_t stream)
{
  (void)in_sizes; (void)n_in; (void)out_size; (void)ws_size;
  const float* x   = (const float*)d_in[0];
  const int*   y   = (const int*)d_in[1];
  const float* W1  = (const float*)d_in[2];
  const float* b1  = (const float*)d_in[3];
  const float* Wg2 = (const float*)d_in[4];
  const float* We2 = (const float*)d_in[5];
  const float* be2 = (const float*)d_in[6];
  const float* Wg3 = (const float*)d_in[7];
  const float* We3 = (const float*)d_in[8];
  const float* be3 = (const float*)d_in[9];

  const size_t TD = (size_t)TOKENS * DMODEL;   // 16M elements
  const size_t DD = (size_t)DMODEL * DMODEL;   // 1M elements
  char* p = (char*)d_ws;
  float*  h32   = (float*)p;   p += TD * 4;
  float*  eo    = (float*)p;   p += TD * 4;
  float*  o32   = (float*)p;   p += TD * 4;
  __bf16* xb    = (__bf16*)p;  p += TD * 2;
  __bf16* dispb = (__bf16*)p;  p += TD * 2;
  __bf16* W1t   = (__bf16*)p;  p += DD * 2;
  __bf16* We2t  = (__bf16*)p;  p += NEXP * DD * 2;
  __bf16* We3t  = (__bf16*)p;  p += NEXP * DD * 2;
  int*    idxb  = (int*)p;     p += TOKENS * 4;
  int*    slotb = (int*)p;     p += TOKENS * 4;
  float*  gateb = (float*)p;   p += TOKENS * 4;
  float*  sent  = (float*)p;

  // ---- one-time operand preparation (outside all GEMM hot loops) ----
  f32_to_bf16_kernel<<<TD / 8 / 256, 256, 0, stream>>>(x, xb);
  transpose_w_kernel<<<dim3(32, 32, 1),    256, 0, stream>>>(W1,  W1t);
  transpose_w_kernel<<<dim3(32, 32, NEXP), 256, 0, stream>>>(We2, We2t);
  transpose_w_kernel<<<dim3(32, 32, NEXP), 256, 0, stream>>>(We3, We3t);

  // Dense pre-layer: h = x @ W1 + b1
  gemm_bf16_kernel<<<dim3(8, TOKENS / 128, 1), 256, 0, stream>>>(
      xb, W1t, b1, h32, 0LL, 0LL, 0LL, 0LL);
  // MoE layer 2 (h32 -> o32) and layer 3 (o32 -> o32)
  run_moe_layer(h32, Wg2, We2t, be2, dispb, eo, o32, idxb, slotb, gateb, stream);
  run_moe_layer(o32, Wg3, We3t, be3, dispb, eo, o32, idxb, slotb, gateb, stream);
  // Residual + sequence mean + cross-entropy
  mean_kernel<<<8, 1024, 0, stream>>>(h32, o32, sent);
  loss_kernel<<<1, 1024, 0, stream>>>(sent, y, (float*)d_out);
}
